// TrigramWriteMemoryAttention_15109694947390
// MI455X (gfx1250) — matrix operand order repaired and checked by reading, not compiled, not run on hardware
//
#include <hip/hip_runtime.h>
#include <hip/hip_bf16.h>

// ---------------------------------------------------------------------------
// Types for CDNA5 WMMA (wave32). bf16 operands, f32 accumulate.
// ---------------------------------------------------------------------------
typedef __attribute__((ext_vector_type(16))) __bf16 v16bf;
typedef __attribute__((ext_vector_type(8)))  float  v8f;
typedef __attribute__((ext_vector_type(8)))  unsigned short u16x8;
typedef __attribute__((ext_vector_type(4)))  unsigned short u16x4;
typedef __attribute__((ext_vector_type(4)))  unsigned int   u32x4;
typedef __attribute__((ext_vector_type(8)))  int            i32x8;
typedef __attribute__((ext_vector_type(4)))  int            i32x4;

union Frag16 {
    v16bf v;
    u16x8 h[2];
};

__device__ __forceinline__ unsigned short f32_to_bf16(float f) {
    union { float f; unsigned int u; } x;
    x.f = f;
    unsigned int lsb = (x.u >> 16) & 1u;
    unsigned int r = x.u + 0x7fffu + lsb;   // round to nearest even
    return (unsigned short)(r >> 16);
}

// A-matrix 16x32 bf16 fragment. LDS holds [M][K] row-major, ld multiple of 8.
// ISA 7.12.2: lanes 0-15 (rows 0..15) hold K kb+0..7 then 16+kb..16+kb+7 with
// kb=0; lanes 16-31 same rows with kb=8.  Two contiguous 16B runs -> 2x b128.
__device__ __forceinline__ v16bf lds_load_Amaj(const unsigned short* s, int ld,
                                               int row0, int k0) {
    int lane = threadIdx.x & 31;
    const unsigned short* base =
        s + (row0 + (lane & 15)) * ld + k0 + ((lane & 16) ? 8 : 0);
    Frag16 f;
    f.h[0] = *(const u16x8*)(base);
    f.h[1] = *(const u16x8*)(base + 16);
    return f.v;
}

// B-matrix 32x16 bf16 fragment. LDS holds the tile TRANSPOSED, [N][K]
// row-major (ld multiple of 8). Lane n = col0+(lane&15); elements are
// K kb+0..7 then 16+kb..16+kb+7 (kb = 0 or 8) -> two 16B runs -> 2x b128.
__device__ __forceinline__ v16bf lds_load_Bmaj(const unsigned short* s, int ld,
                                               int col0, int k0) {
    int lane = threadIdx.x & 31;
    const unsigned short* base =
        s + (col0 + (lane & 15)) * ld + k0 + ((lane & 16) ? 8 : 0);
    Frag16 f;
    f.h[0] = *(const u16x8*)(base);
    f.h[1] = *(const u16x8*)(base + 16);
    return f.v;
}

__device__ __forceinline__ v8f wmma_bf16(v16bf a, v16bf b, v8f c) {
    return __builtin_amdgcn_wmma_f32_16x16x32_bf16(false, a, false, b,
                                                   (short)0, c, false, false);
}

// ---------------------------------------------------------------------------
// TDM: DMA a 2D tile (tileM rows x tileK elements, 2-byte elements, row
// stride rowStride elements) from global memory into LDS at lds_byte, with
// LDS padding of 16B after every 64B row => 80B (40-ushort) LDS pitch.
// D# layout per CDNA5 ISA ch.8 (group0 128b, group1 256b; groups 2/3 unused
// for 2D). Issued once per wave; completion tracked with TENSORcnt.
// ---------------------------------------------------------------------------
__device__ __forceinline__ void tdm_load_tile_2d(const void* gptr,
                                                 unsigned lds_byte,
                                                 int rowStride, int numRows,
                                                 int tileK, int tileM) {
    unsigned long long ga = (unsigned long long)(uintptr_t)gptr;
    u32x4 g0;
    g0[0] = 1u;                                    // count=1, user mode
    g0[1] = lds_byte;                              // lds_addr [63:32]
    g0[2] = (unsigned)(ga & 0xFFFFFFFFu);          // global_addr [95:64]
    g0[3] = (unsigned)((ga >> 32) & 0x01FFFFFFu)   // global_addr [120:96]
            | (2u << 30);                          // type=2 (image)
    i32x8 g1;
    // data_size=1 (2B), pad_enable, pad_interval=3 (16 DW), pad_amount=3 (4 DW)
    g1[0] = (int)((1u << 16) | (1u << 20) | (3u << 22) | (3u << 25));
    unsigned td0 = (unsigned)rowStride;            // tensor_dim0  [79:48]
    unsigned td1 = (unsigned)numRows;              // tensor_dim1  [111:80]
    g1[1] = (int)((td0 & 0xFFFFu) << 16);
    g1[2] = (int)((td0 >> 16) | ((td1 & 0xFFFFu) << 16));
    g1[3] = (int)((td1 >> 16) | ((unsigned)tileK << 16));   // tile_dim0
    g1[4] = (int)((unsigned)tileM);                // tile_dim1, tile_dim2=0
    g1[5] = rowStride;                             // tensor_dim0_stride lo32
    g1[6] = 0;                                     // stride hi, dim1_stride lo
    g1[7] = 0;
    i32x4 z4 = {0, 0, 0, 0};
    i32x8 z8 = {0, 0, 0, 0, 0, 0, 0, 0};
    __builtin_amdgcn_tensor_load_to_lds(g0, g1, z4, z4, z8, 0);
}

// ---------------------------------------------------------------------------
// Generic GEMM: C[M,N] (f32) = A[M,K](bf16) @ B[K,N](bf16) (+ bias[N]).
// 256 threads = 8 waves; block tile 128x128, wave tile 32x64, K-step 32.
// A tile: TDM (tensor_load_to_lds) into double-buffered LDS, issued by wave 0
// and overlapped with compute (wait via s_wait_tensorcnt before barrier).
// B tile: register-staged + transposed into LDS ([n][k]) for vector B-frags.
// M % 128 == 0, K % 32 == 0, N % 8 == 0 and guarded (N = 16/64/1024/3072).
// ---------------------------------------------------------------------------
__global__ __launch_bounds__(256) void gemm_bf16_kernel(
    const unsigned short* __restrict__ A, const unsigned short* __restrict__ B,
    const float* __restrict__ bias, float* __restrict__ C, int M, int N, int K) {
    __shared__ unsigned short As[2][128 * 40];   // [m][k], 80B pitch via TDM pad
    __shared__ unsigned short Bs[2][128 * 40];   // transposed: [n][k]
    const int tid = threadIdx.x;
    const int wave = tid >> 5;
    const int lane = tid & 31;
    const int mw = (wave & 3) * 32;      // wave row group within tile
    const int nw = (wave >> 2) * 64;     // wave col half within tile
    const int m0 = blockIdx.y * 128;
    const int n0 = blockIdx.x * 128;

    v8f acc[2][4];
#pragma unroll
    for (int mi = 0; mi < 2; ++mi)
#pragma unroll
        for (int j = 0; j < 4; ++j)
#pragma unroll
            for (int e = 0; e < 8; ++e) acc[mi][j][e] = 0.f;

    u16x8 bR[2];

    // B: 32x128 = 512 chunks of 8 ushorts; 2 per thread.
    auto gloadB = [&](int k0) {
#pragma unroll
        for (int q = 0; q < 2; ++q) {
            int i = tid + q * 256;
            int k = i >> 4, n8 = (i & 15) * 8;
            if (n0 + n8 + 7 < N) {
                bR[q] = *(const u16x8*)&B[(size_t)(k0 + k) * N + n0 + n8];
            } else {
#pragma unroll
                for (int j2 = 0; j2 < 8; ++j2)
                    bR[q][j2] = (n0 + n8 + j2 < N)
                                    ? B[(size_t)(k0 + k) * N + n0 + n8 + j2]
                                    : (unsigned short)0;
            }
        }
        if (k0 + 64 < K)
            __builtin_prefetch(&B[(size_t)(k0 + 32 + (tid >> 4)) * N + n0], 0, 0);
    };
    auto lstoreB = [&](int p) {
        unsigned short* bs = Bs[p];
#pragma unroll
        for (int q = 0; q < 2; ++q) {
            int i = tid + q * 256;
            int k = i >> 4, n8 = (i & 15) * 8;
#pragma unroll
            for (int j2 = 0; j2 < 8; ++j2)
                bs[(n8 + j2) * 40 + k] = bR[q][j2];
        }
    };
    auto tdmA = [&](int k0, int p) {
        tdm_load_tile_2d(&A[(size_t)m0 * K + k0],
                         (unsigned)(uintptr_t)&As[p][0], K, M - m0, 32, 128);
    };

    const int KT = K >> 5;
    gloadB(0);
    if (wave == 0) tdmA(0, 0);
    for (int kt = 0; kt < KT; ++kt) {
        const int p = kt & 1;
        lstoreB(p);
        if (wave == 0) __builtin_amdgcn_s_wait_tensorcnt(0);
        __syncthreads();
        if (kt + 1 < KT) {                       // overlap next tile with compute
            gloadB((kt + 1) << 5);
            if (wave == 0) tdmA((kt + 1) << 5, 1 - p);
        }
        const unsigned short* as = As[p];
        const unsigned short* bs = Bs[p];
        v16bf a0 = lds_load_Amaj(as, 40, mw, 0);
        v16bf a1 = lds_load_Amaj(as, 40, mw + 16, 0);
#pragma unroll
        for (int j = 0; j < 4; ++j) {
            v16bf b = lds_load_Bmaj(bs, 40, nw + j * 16, 0);
            acc[0][j] = wmma_bf16(a0, b, acc[0][j]);
            acc[1][j] = wmma_bf16(a1, b, acc[1][j]);
        }
    }

    const int coll = lane & 15;
    const int rowadd = (lane & 16) ? 8 : 0;
#pragma unroll
    for (int mi = 0; mi < 2; ++mi)
#pragma unroll
        for (int j = 0; j < 4; ++j) {
            int n = n0 + nw + j * 16 + coll;
            if (n >= N) continue;
            float bv = bias ? bias[n] : 0.f;
#pragma unroll
            for (int r = 0; r < 8; ++r) {
                int m = m0 + mw + mi * 16 + r + rowadd;
                C[(size_t)m * N + n] = acc[mi][j][r] + bv;
            }
        }
}

// ---------------------------------------------------------------------------
// Flash attention, causal, per (b, h, 64-row q tile). dh = 64, T = 1024.
// 128 threads = 4 waves, wave w owns rows [16w, 16w+16).
// ---------------------------------------------------------------------------
__global__ __launch_bounds__(128) void flash_attn_kernel(
    const float* __restrict__ qkv, float* __restrict__ out, int B, int T, int H) {
    const int dh = 64;
    const int qt = blockIdx.x;
    const int bh = blockIdx.y;
    const int b = bh / H, h = bh % H;
    const int t0 = qt * 64;
    const int tid = threadIdx.x, wave = tid >> 5, lane = tid & 31;
    const int D3 = 3 * H * dh;            // 3072
    const int D = H * dh;                 // 1024
    const float scale = 0.125f;           // dh^-0.5

    __shared__ unsigned short Qs[64 * 72];   // [m][k]   (row-major)
    __shared__ unsigned short Ks[64 * 72];   // [s][dh]  == [n][k] for Q@K^T
    __shared__ unsigned short Vs[64 * 72];   // transposed: [dh][s] == [n][k]
    __shared__ unsigned short Ps[64 * 72];   // [m][s]   (row-major)

    // load Q tile (scaled), vectorized float4 -> ushort4
    for (int i = tid; i < 64 * 16; i += 128) {
        int r = i >> 4, c4 = (i & 15) * 4;
        float4 q = *(const float4*)&qkv[(size_t)(b * T + t0 + r) * D3 +
                                        h * dh + c4];
        u16x4 o = {f32_to_bf16(q.x * scale), f32_to_bf16(q.y * scale),
                   f32_to_bf16(q.z * scale), f32_to_bf16(q.w * scale)};
        *(u16x4*)&Qs[r * 72 + c4] = o;
    }
    __syncthreads();

    v8f o[4];
    float mrow[8], lrow[8];
#pragma unroll
    for (int j = 0; j < 4; ++j)
#pragma unroll
        for (int e = 0; e < 8; ++e) o[j][e] = 0.f;
#pragma unroll
    for (int r = 0; r < 8; ++r) { mrow[r] = -3.0e38f; lrow[r] = 0.f; }

    const int coll = lane & 15;
    const int rowadd = (lane & 16) ? 8 : 0;

    for (int j = 0; j <= qt; ++j) {
        const int s0 = j * 64;
        // stage K row-major, V transposed
        for (int i = tid; i < 64 * 16; i += 128) {
            int r = i >> 4, c4 = (i & 15) * 4;
            size_t base = (size_t)(b * T + s0 + r) * D3 + h * dh + c4;
            float4 kv = *(const float4*)&qkv[base + D];
            float4 vv = *(const float4*)&qkv[base + 2 * D];
            u16x4 ko = {f32_to_bf16(kv.x), f32_to_bf16(kv.y),
                        f32_to_bf16(kv.z), f32_to_bf16(kv.w)};
            *(u16x4*)&Ks[r * 72 + c4] = ko;
            Vs[(c4 + 0) * 72 + r] = f32_to_bf16(vv.x);
            Vs[(c4 + 1) * 72 + r] = f32_to_bf16(vv.y);
            Vs[(c4 + 2) * 72 + r] = f32_to_bf16(vv.z);
            Vs[(c4 + 3) * 72 + r] = f32_to_bf16(vv.w);
        }
        __syncthreads();

        // S = Q @ K^T  (scale folded into Q)
        v8f s[4];
#pragma unroll
        for (int jc = 0; jc < 4; ++jc)
#pragma unroll
            for (int e = 0; e < 8; ++e) s[jc][e] = 0.f;
#pragma unroll
        for (int k0 = 0; k0 < 64; k0 += 32) {
            v16bf a = lds_load_Amaj(Qs, 72, wave * 16, k0);
#pragma unroll
            for (int jc = 0; jc < 4; ++jc) {
                v16bf bm = lds_load_Bmaj(Ks, 72, jc * 16, k0);
                s[jc] = wmma_bf16(a, bm, s[jc]);
            }
        }

        if (j == qt) {  // causal mask on diagonal tile
#pragma unroll
            for (int jc = 0; jc < 4; ++jc) {
                int n = s0 + jc * 16 + coll;
#pragma unroll
                for (int r = 0; r < 8; ++r) {
                    int m = t0 + wave * 16 + r + rowadd;
                    if (n > m) s[jc][r] = -3.0e38f;
                }
            }
        }

        // online softmax, per row (row r / r+8 stats stay lane-aligned)
#pragma unroll
        for (int r = 0; r < 8; ++r) {
            float mx = -3.0e38f;
#pragma unroll
            for (int jc = 0; jc < 4; ++jc) mx = fmaxf(mx, s[jc][r]);
#pragma unroll
            for (int off = 8; off >= 1; off >>= 1)
                mx = fmaxf(mx, __shfl_xor(mx, off, 32));
            float mnew = fmaxf(mrow[r], mx);
            float corr = __expf(mrow[r] - mnew);
            float rs = 0.f;
#pragma unroll
            for (int jc = 0; jc < 4; ++jc) {
                float p = __expf(s[jc][r] - mnew);
                s[jc][r] = p;
                rs += p;
            }
#pragma unroll
            for (int off = 8; off >= 1; off >>= 1)
                rs += __shfl_xor(rs, off, 32);
            lrow[r] = lrow[r] * corr + rs;
            mrow[r] = mnew;
#pragma unroll
            for (int jc = 0; jc < 4; ++jc) o[jc][r] *= corr;
        }

        // P (C-layout) -> LDS row-major -> A-layout
#pragma unroll
        for (int jc = 0; jc < 4; ++jc)
#pragma unroll
            for (int r = 0; r < 8; ++r)
                Ps[(wave * 16 + r + rowadd) * 72 + jc * 16 + coll] =
                    f32_to_bf16(s[jc][r]);

        // O += P @ V
#pragma unroll
        for (int k0 = 0; k0 < 64; k0 += 32) {
            v16bf a = lds_load_Amaj(Ps, 72, wave * 16, k0);
#pragma unroll
            for (int jc = 0; jc < 4; ++jc) {
                v16bf bm = lds_load_Bmaj(Vs, 72, jc * 16, k0);
                o[jc] = wmma_bf16(a, bm, o[jc]);
            }
        }
        __syncthreads();
    }

#pragma unroll
    for (int r = 0; r < 8; ++r) {
        float inv = 1.f / lrow[r];
        int m = t0 + wave * 16 + r + rowadd;
#pragma unroll
        for (int jc = 0; jc < 4; ++jc) {
            int c = jc * 16 + coll;
            out[(size_t)(b * T + m) * D + h * dh + c] = o[jc][r] * inv;
        }
    }
}

// ---------------------------------------------------------------------------
// Elementwise helpers
// ---------------------------------------------------------------------------
// n4 = element count / 4
__global__ void cvt_bf16_kernel(const float* __restrict__ s,
                                unsigned short* __restrict__ d, size_t n4) {
    size_t i = blockIdx.x * (size_t)blockDim.x + threadIdx.x;
    if (i >= n4) return;
    float4 v = *(const float4*)&s[i * 4];
    u16x4 o = {f32_to_bf16(v.x), f32_to_bf16(v.y), f32_to_bf16(v.z),
               f32_to_bf16(v.w)};
    *(u16x4*)&d[i * 4] = o;
}

// ctx[bt][0:D] = x[b][t-2], ctx[bt][D:2D] = x[b][t-1]  (zeros at edges)
__global__ void build_ctx_kernel(const float* __restrict__ x,
                                 unsigned short* __restrict__ ctx, int B, int T,
                                 int D) {
    size_t i = blockIdx.x * (size_t)blockDim.x + threadIdx.x;  // float4 chunks
    size_t total = (size_t)B * T * 2 * D / 4;
    if (i >= total) return;
    int col = (int)((i * 4) % (2 * D));
    size_t bt = (i * 4) / (2 * D);
    int b = (int)(bt / T), t = (int)(bt % T);
    int shift = (col < D) ? 2 : 1;
    int c = (col < D) ? col : col - D;
    u16x4 o = {0, 0, 0, 0};
    if (t - shift >= 0) {
        float4 v = *(const float4*)&x[((size_t)b * T + t - shift) * D + c];
        o[0] = f32_to_bf16(v.x);
        o[1] = f32_to_bf16(v.y);
        o[2] = f32_to_bf16(v.z);
        o[3] = f32_to_bf16(v.w);
    }
    *(u16x4*)&ctx[i * 4] = o;
}

// Plucker exterior product, normalize, optionally apply J permutation.
__global__ void exterior_kernel(const float* __restrict__ P1,
                                const float* __restrict__ P2,
                                float* __restrict__ Lout, int applyJ, int BT,
                                int H) {
    int i = blockIdx.x * blockDim.x + threadIdx.x;
    if (i >= BT * H) return;
    int bt = i / H, h = i % H;
    const float4 a = *(const float4*)&P1[((size_t)bt * H + h) * 4];
    const float4 b = *(const float4*)&P2[((size_t)bt * H + h) * 4];
    float L0 = a.x * b.y - a.y * b.x;
    float L1 = a.x * b.z - a.z * b.x;
    float L2 = a.x * b.w - a.w * b.x;
    float L3 = a.y * b.z - a.z * b.y;
    float L4 = a.y * b.w - a.w * b.y;
    float L5 = a.z * b.w - a.w * b.z;
    float n2 = L0 * L0 + L1 * L1 + L2 * L2 + L3 * L3 + L4 * L4 + L5 * L5;
    float inv = 1.f / fmaxf(sqrtf(n2), 1e-12f);
    float* o = &Lout[(size_t)i * 6];
    if (applyJ) {   // L @ J  with J = antidiag(+1,-1,+1,+1,-1,+1)
        o[0] =  L5 * inv; o[1] = -L4 * inv; o[2] =  L3 * inv;
        o[3] =  L2 * inv; o[4] = -L1 * inv; o[5] =  L0 * inv;
    } else {
        o[0] = L0 * inv; o[1] = L1 * inv; o[2] = L2 * inv;
        o[3] = L3 * inv; o[4] = L4 * inv; o[5] = L5 * inv;
    }
}

// mem_score[b,h,t] = sum_{s<t} decay^(t-s) * (r_t . Jw_s)^2
__global__ __launch_bounds__(128) void memscore_kernel(
    const float* __restrict__ RL, const float* __restrict__ JW,
    const float* __restrict__ dlog, float* __restrict__ ms, int B, int T,
    int H) {
    const int bh = blockIdx.y;
    const int b = bh / H, h = bh % H;
    const int t = blockIdx.x * 128 + threadIdx.x;
    float r[6];
#pragma unroll
    for (int i = 0; i < 6; ++i)
        r[i] = RL[(((size_t)(b * T + t)) * H + h) * 6 + i];
    const float d = 1.f / (1.f + __expf(-dlog[h]));
    const float ld = __logf(d);
    __shared__ float Jt[128 * 6];
    float acc = 0.f;
    for (int s0 = blockIdx.x * 128; s0 >= 0; s0 -= 128) {
        for (int i = threadIdx.x; i < 128 * 6; i += 128)
            Jt[i] = JW[(((size_t)(b * T + s0 + i / 6)) * H + h) * 6 + (i % 6)];
        __syncthreads();
        int sm = s0 + 127 < t - 1 ? s0 + 127 : t - 1;
        if (sm >= s0) {
            float f = __expf(ld * (float)(t - sm));
            for (int s = sm; s >= s0; --s) {
                const float* w = &Jt[(s - s0) * 6];
                float dot = r[0] * w[0] + r[1] * w[1] + r[2] * w[2] +
                            r[3] * w[3] + r[4] * w[4] + r[5] * w[5];
                acc += f * dot * dot;
                f *= d;
            }
        }
        __syncthreads();
    }
    ms[(size_t)bh * T + t] = acc;
}

// gated[bt] = mean_h sigmoid(ms*scale_h) * sigmoid(gate_pre)
__global__ void gated_kernel(const float* __restrict__ ms,
                             const float* __restrict__ gatep,
                             const float* __restrict__ mem_scale,
                             float* __restrict__ gated, int B, int T, int H) {
    int bt = blockIdx.x * blockDim.x + threadIdx.x;
    if (bt >= B * T) return;
    int b = bt / T, t = bt % T;
    float acc = 0.f;
    for (int h = 0; h < H; ++h) {
        float m = ms[((size_t)(b * H + h)) * T + t];
        float g1 = 1.f / (1.f + __expf(-m * mem_scale[h]));
        float g2 = 1.f / (1.f + __expf(-gatep[(size_t)bt * H + h]));
        acc += g1 * g2;
    }
    gated[bt] = acc / (float)H;
}

// y = seq_out + gated * mem_val  ->  bf16 (vectorized by 4)
__global__ void combine_kernel(const float* __restrict__ seq,
                               const float* __restrict__ memval,
                               const float* __restrict__ gated,
                               unsigned short* __restrict__ ybf, int D,
                               size_t n4) {
    size_t i = blockIdx.x * (size_t)blockDim.x + threadIdx.x;
    if (i >= n4) return;
    size_t bt = (i * 4) / D;
    float g = gated[bt];
    float4 sv = *(const float4*)&seq[i * 4];
    float4 mv = *(const float4*)&memval[i * 4];
    u16x4 o = {f32_to_bf16(sv.x + g * mv.x), f32_to_bf16(sv.y + g * mv.y),
               f32_to_bf16(sv.z + g * mv.z), f32_to_bf16(sv.w + g * mv.w)};
    *(u16x4*)&ybf[i * 4] = o;
}

// ---------------------------------------------------------------------------
// Host launch
// ---------------------------------------------------------------------------
static inline size_t align256(size_t x) { return (x + 255) & ~(size_t)255; }

extern "C" void kernel_launch(void* const* d_in, const int* in_sizes, int n_in,
                              void* d_out, int out_size, void* d_ws,
                              size_t ws_size, hipStream_t stream) {
    const int B = 2, T = 1024, D = 1024, H = 16;
    const int BT = B * T;          // 2048
    (void)in_sizes; (void)n_in; (void)out_size; (void)ws_size;

    const float* x            = (const float*)d_in[0];
    const float* qkv_w        = (const float*)d_in[1];
    const float* qkv_b        = (const float*)d_in[2];
    const float* w1_write_w   = (const float*)d_in[3];
    const float* w2_write_w   = (const float*)d_in[4];
    const float* w1_read_w    = (const float*)d_in[5];
    const float* w2_read_w    = (const float*)d_in[6];
    const float* mem_value_w  = (const float*)d_in[7];
    const float* mem_value_b  = (const float*)d_in[8];
    const float* mem_gate_w   = (const float*)d_in[9];
    const float* mem_gate_b   = (const float*)d_in[10];
    const float* mem_scale    = (const float*)d_in[11];
    const float* out_w        = (const float*)d_in[12];
    const float* out_b        = (const float*)d_in[13];
    const float* decay_logits = (const float*)d_in[14];

    // workspace carve-up
    char* p = (char*)d_ws;
    auto alloc = [&](size_t bytes) -> void* {
        void* r = (void*)p;
        p += align256(bytes);
        return r;
    };
    unsigned short* xb16   = (unsigned short*)alloc((size_t)BT * D * 2);
    unsigned short* ctx    = (unsigned short*)alloc((size_t)BT * 2 * D * 2);
    unsigned short* wqkv   = (unsigned short*)alloc((size_t)D * 3 * D * 2);
    unsigned short* w1w    = (unsigned short*)alloc((size_t)2 * D * 64 * 2);
    unsigned short* w2w    = (unsigned short*)alloc((size_t)D * 64 * 2);
    unsigned short* r1w    = (unsigned short*)alloc((size_t)D * 64 * 2);
    unsigned short* r2w    = (unsigned short*)alloc((size_t)D * 64 * 2);
    unsigned short* mvw    = (unsigned short*)alloc((size_t)D * D * 2);
    unsigned short* mgw    = (unsigned short*)alloc((size_t)D * H * 2);
    unsigned short* ow     = (unsigned short*)alloc((size_t)D * D * 2);
    float* qkvf   = (float*)alloc((size_t)BT * 3 * D * 4);
    float* seqout = (float*)alloc((size_t)BT * D * 4);
    float* w1f    = (float*)alloc((size_t)BT * 64 * 4);
    float* w2f    = (float*)alloc((size_t)BT * 64 * 4);
    float* r1f    = (float*)alloc((size_t)BT * 64 * 4);
    float* r2f    = (float*)alloc((size_t)BT * 64 * 4);
    float* JW     = (float*)alloc((size_t)BT * H * 6 * 4);
    float* RL     = (float*)alloc((size_t)BT * H * 6 * 4);
    float* msb    = (float*)alloc((size_t)B * H * T * 4);
    float* memval = (float*)alloc((size_t)BT * D * 4);
    float* gatep  = (float*)alloc((size_t)BT * H * 4);
    float* gated  = (float*)alloc((size_t)BT * 4);
    unsigned short* ybf = (unsigned short*)alloc((size_t)BT * D * 2);

    auto cvt = [&](const float* src, unsigned short* dst, size_t n) {
        size_t n4 = n / 4;
        cvt_bf16_kernel<<<(unsigned)((n4 + 255) / 256), 256, 0, stream>>>(src,
                                                                          dst,
                                                                          n4);
    };

    // bf16 staging
    cvt(x, xb16, (size_t)BT * D);
    build_ctx_kernel<<<(unsigned)(((size_t)BT * 2 * D / 4 + 255) / 256), 256, 0,
                      stream>>>(x, ctx, B, T, D);
    cvt(qkv_w, wqkv, (size_t)D * 3 * D);
    cvt(w1_write_w, w1w, (size_t)2 * D * 64);
    cvt(w2_write_w, w2w, (size_t)D * 64);
    cvt(w1_read_w, r1w, (size_t)D * 64);
    cvt(w2_read_w, r2w, (size_t)D * 64);
    cvt(mem_value_w, mvw, (size_t)D * D);
    cvt(mem_gate_w, mgw, (size_t)D * H);
    cvt(out_w, ow, (size_t)D * D);

    // qkv = x @ qkv_w + b   (M=2048, N=3072, K=1024)
    gemm_bf16_kernel<<<dim3(24, 16), 256, 0, stream>>>(xb16, wqkv, qkv_b, qkvf,
                                                       BT, 3 * D, D);
    // causal attention
    flash_attn_kernel<<<dim3(16, 32), 128, 0, stream>>>(qkvf, seqout, B, T, H);

    // point projections (N=64)
    gemm_bf16_kernel<<<dim3(1, 16), 256, 0, stream>>>(ctx, w1w, nullptr, w1f,
                                                      BT, 64, 2 * D);
    gemm_bf16_kernel<<<dim3(1, 16), 256, 0, stream>>>(xb16, w2w, nullptr, w2f,
                                                      BT, 64, D);
    gemm_bf16_kernel<<<dim3(1, 16), 256, 0, stream>>>(xb16, r1w, nullptr, r1f,
                                                      BT, 64, D);
    gemm_bf16_kernel<<<dim3(1, 16), 256, 0, stream>>>(xb16, r2w, nullptr, r2f,
                                                      BT, 64, D);

    // Plucker lines (write side gets J applied)
    exterior_kernel<<<(BT * H + 255) / 256, 256, 0, stream>>>(w1f, w2f, JW, 1,
                                                              BT, H);
    exterior_kernel<<<(BT * H + 255) / 256, 256, 0, stream>>>(r1f, r2f, RL, 0,
                                                              BT, H);

    // decayed incidence score
    memscore_kernel<<<dim3(8, 32), 128, 0, stream>>>(RL, JW, decay_logits, msb,
                                                     B, T, H);

    // gate (N=16) + memory value (N=1024)
    gemm_bf16_kernel<<<dim3(1, 16), 256, 0, stream>>>(xb16, mgw, mem_gate_b,
                                                      gatep, BT, H, D);
    gemm_bf16_kernel<<<dim3(8, 16), 256, 0, stream>>>(xb16, mvw, mem_value_b,
                                                      memval, BT, D, D);
    gated_kernel<<<(BT + 255) / 256, 256, 0, stream>>>(msb, gatep, mem_scale,
                                                       gated, B, T, H);
    combine_kernel<<<(unsigned)(((size_t)BT * D / 4 + 255) / 256), 256, 0,
                    stream>>>(seqout, memval, gated, ybf, D, (size_t)BT * D / 4);

    // output projection -> d_out  (N=1024)
    gemm_bf16_kernel<<<dim3(8, 16), 256, 0, stream>>>(ybf, ow, out_b,
                                                      (float*)d_out, BT, D, D);
}
